// TalkingHeadAttention_4020089389491
// MI455X (gfx1250) — matrix operand order repaired and checked by reading, not compiled, not run on hardware
//
#include <hip/hip_runtime.h>
#include <hip/hip_bf16.h>

typedef __attribute__((ext_vector_type(16))) __bf16 v16bf;
typedef __attribute__((ext_vector_type(8)))  __bf16 v8bf;
typedef __attribute__((ext_vector_type(8)))  float  v8f;
typedef __attribute__((ext_vector_type(4)))  float  v4f;

// ---------------------------------------------------------------------------
// WMMA helper: D = A(16x32 bf16) * B(32x16 bf16) + C(16x16 f32)
// ---------------------------------------------------------------------------
__device__ __forceinline__ v8f wmma_bf16(v16bf a, v16bf b, v8f c) {
  return __builtin_amdgcn_wmma_f32_16x16x32_bf16(false, a, false, b,
                                                 (short)0, c, false, false);
}

// A operand layout (16-bit A 16x32, ISA 7.12.2):
//   lane<16 : row M=lane,    slots 0..7 -> K=0..7,  slots 8..15 -> K=16..23
//   lane>=16: row M=lane-16, slots 0..7 -> K=8..15, slots 8..15 -> K=24..31
__device__ __forceinline__ v16bf loadA_bf16(const __bf16* tile, int stride, int lane) {
  const int half = lane >> 4, m = lane & 15;
  const __bf16* row = tile + m * stride;
  v8bf lo = *(const v8bf*)(row + (half ? 8 : 0));
  v8bf hi = *(const v8bf*)(row + (half ? 24 : 16));
  v16bf a;
#pragma unroll
  for (int i = 0; i < 8; ++i) { a[i] = lo[i]; a[i + 8] = hi[i]; }
  return a;
}

__device__ __forceinline__ v16bf loadA_f32(const float* tile, int stride, int lane) {
  const int half = lane >> 4, m = lane & 15;
  const float* row = tile + m * stride;
  const int k0 = half ? 8 : 0, k1 = half ? 24 : 16;
  v4f f0 = *(const v4f*)(row + k0);
  v4f f1 = *(const v4f*)(row + k0 + 4);
  v4f f2 = *(const v4f*)(row + k1);
  v4f f3 = *(const v4f*)(row + k1 + 4);
  v16bf a;
#pragma unroll
  for (int i = 0; i < 4; ++i) {
    a[i]      = (__bf16)f0[i];
    a[i + 4]  = (__bf16)f1[i];
    a[i + 8]  = (__bf16)f2[i];
    a[i + 12] = (__bf16)f3[i];
  }
  return a;
}

// B operand layout (16-bit B 32x16): lane holds column N=lane%16,
// slots 0..7 -> K = (lane/16)*8 + i, slots 8..15 -> K = 16 + (lane/16)*8 + (i-8).
// `tile` points at B element (K=0, N=0); strideN = distance between columns.
__device__ __forceinline__ v16bf loadB_bf16(const __bf16* tile, int strideN, int lane) {
  const int half = lane >> 4, n = lane & 15;
  const __bf16* col = tile + n * strideN;
  v8bf lo = *(const v8bf*)(col + (half ? 8 : 0));
  v8bf hi = *(const v8bf*)(col + (half ? 24 : 16));
  v16bf b;
#pragma unroll
  for (int i = 0; i < 8; ++i) { b[i] = lo[i]; b[i + 8] = hi[i]; }
  return b;
}

__device__ __forceinline__ v16bf loadB_f32(const float* tile, int strideN, int lane) {
  const int half = lane >> 4, n = lane & 15;
  const float* col = tile + n * strideN;
  const int k0 = half ? 8 : 0, k1 = half ? 24 : 16;
  v4f f0 = *(const v4f*)(col + k0);
  v4f f1 = *(const v4f*)(col + k0 + 4);
  v4f f2 = *(const v4f*)(col + k1);
  v4f f3 = *(const v4f*)(col + k1 + 4);
  v16bf b;
#pragma unroll
  for (int i = 0; i < 4; ++i) {
    b[i]      = (__bf16)f0[i];
    b[i + 4]  = (__bf16)f1[i];
    b[i + 8]  = (__bf16)f2[i];
    b[i + 12] = (__bf16)f3[i];
  }
  return b;
}

// ---------------------------------------------------------------------------
// Problem constants
// ---------------------------------------------------------------------------
#define NB    2
#define NSEQ  2048
#define DIM   512
#define HEADS 8
#define HD    64
#define QCH   256          // rope chunk
// ln(10000)/16
#define ROPE_LOG 0.5756462732485114f

// ===========================================================================
// Kernel 1: qkv = x @ w_qkv^T + b ; RoPE(q,k) -> bf16 [b][h][n][64];
//           v -> bf16 transposed [b][h][64][n]
// wave computes a 16x64 output tile; 24 column groups = (3 mats x 8 heads)
// ===========================================================================
__global__ __launch_bounds__(256) void k_qkv_rope(
    const float* __restrict__ x, const float* __restrict__ wqkv,
    const float* __restrict__ bqkv,
    __bf16* __restrict__ qb, __bf16* __restrict__ kb, __bf16* __restrict__ vT) {
  const int lane = threadIdx.x & 31;
  const int w = blockIdx.x * 8 + (threadIdx.x >> 5);    // 6144 waves
  const int rowTile = w / 24;                           // 256 tiles of 16 rows
  const int cg = w - rowTile * 24;                      // 0..23
  const int which = cg >> 3, head = cg & 7;

  const float* A0 = x + (rowTile * 16) * DIM;
  const float* B0 = wqkv + (cg * 64) * DIM;             // rows of w_qkv = B cols

  v8f acc[4] = {};
  for (int k = 0; k < DIM; k += 32) {
    if (k + 32 < DIM) {
      // speculative prefetch of next K panel (one line per lane)
      __builtin_prefetch(A0 + k + 32 + (lane & 15) * DIM, 0, 1);
      __builtin_prefetch(B0 + k + 32 + lane * DIM, 0, 1);
      __builtin_prefetch(B0 + k + 32 + (lane + 32) * DIM, 0, 1);
    }
    v16bf a = loadA_f32(A0 + k, DIM, lane);
#pragma unroll
    for (int t = 0; t < 4; ++t)
      acc[t] = wmma_bf16(a, loadB_f32(B0 + (t * 16) * DIM + k, DIM, lane), acc[t]);
  }

  const int half = lane >> 4, n = lane & 15;
  float bias[4];
#pragma unroll
  for (int t = 0; t < 4; ++t) bias[t] = bqkv[cg * 64 + t * 16 + n];

  if (which < 2) {
    __bf16* dst = (which == 0) ? qb : kb;
    const float inv = __expf(-(float)n * ROPE_LOG);     // 10000^(-n/16)
#pragma unroll
    for (int r = 0; r < 8; ++r) {
      const int gr = rowTile * 16 + r + 8 * half;
      const int bb = gr >> 11, ni = gr & (NSEQ - 1);
      const float x1 = acc[0][r] + bias[0], x2 = acc[1][r] + bias[1];
      const float x3 = acc[2][r] + bias[2], x4 = acc[3][r] + bias[3];
      const float al = (float)(ni & (QCH - 1)) * inv;   // within-chunk angle
      const float ac = (float)(ni >> 8) * inv;          // chunk-index angle
      const float cl = __cosf(al), sl = __sinf(al);
      const float cc = __cosf(ac), sc = __sinf(ac);
      __bf16* o = dst + (((bb * HEADS + head) * NSEQ + ni) * HD);
      o[n]      = (__bf16)(x1 * cl - x2 * sl);
      o[16 + n] = (__bf16)(x1 * sl + x2 * cl);
      o[32 + n] = (__bf16)(x3 * cc - x4 * sc);
      o[48 + n] = (__bf16)(x3 * sc + x4 * cc);
    }
  } else {
#pragma unroll
    for (int r = 0; r < 8; ++r) {
      const int gr = rowTile * 16 + r + 8 * half;
      const int bb = gr >> 11, ni = gr & (NSEQ - 1);
#pragma unroll
      for (int t = 0; t < 4; ++t) {
        const int dcol = t * 16 + n;
        vT[(((bb * HEADS + head) * HD + dcol) * NSEQ) + ni] =
            (__bf16)(acc[t][r] + bias[t]);
      }
    }
  }
}

// ===========================================================================
// Kernel 2: talking-heads attention for one (batch, 16-query tile) per block.
// 8 waves; wave w is head role w in every phase.
//   pass 1: online (max,sum) of pre-mixed logits per mixed head
//   pass 2: recompute logits, normalize, post-mix, PV via WMMA
// ===========================================================================
__global__ __launch_bounds__(256) void k_attn(
    const __bf16* __restrict__ qb, const __bf16* __restrict__ kb,
    const __bf16* __restrict__ vT, const float* __restrict__ wpre,
    const float* __restrict__ wpost, __bf16* __restrict__ ctx) {
  __shared__ __attribute__((aligned(16))) float sL[16 * 32 * 8];  // raw logits [i][j][h]
  __shared__ __attribute__((aligned(16))) float sP[16 * 32 * 8];  // probs      [i][j][g]
  __shared__ __attribute__((aligned(16))) float sMZ[8][16][2];    // (max, 1/sum)

  const int lane = threadIdx.x & 31;
  const int wid = threadIdx.x >> 5;          // head role
  const int bb = blockIdx.x >> 7;
  const int it = blockIdx.x & 127;
  const int i0 = it * 16;

  const __bf16* qh = qb + ((bb * HEADS + wid) * NSEQ + i0) * HD;
  const __bf16* kh = kb + ((bb * HEADS + wid) * NSEQ) * HD;
  const __bf16* vh = vT + ((bb * HEADS + wid) * HD) * NSEQ;

  const v16bf qa0 = loadA_bf16(qh, HD, lane);        // d = 0..31
  const v16bf qa1 = loadA_bf16(qh + 32, HD, lane);   // d = 32..63

  float wpre_r[8], wpost_r[8];
#pragma unroll
  for (int h = 0; h < 8; ++h) {
    wpre_r[h]  = wpre[wid * 8 + h];
    wpost_r[h] = wpost[wid * 8 + h];
  }
  const float scale = 0.125f;                 // 64^-0.5
  const int half = lane >> 4, nn = lane & 15;

  // ---------------- pass 1: softmax stats ----------------
  float rm[16], rs[16];
#pragma unroll
  for (int t = 0; t < 16; ++t) { rm[t] = -1e30f; rs[t] = 0.f; }

  for (int jc = 0; jc < NSEQ / 32; ++jc) {
    const int j0 = jc * 32;
    if (j0 + 32 < NSEQ)  // prefetch next chunk of K (32 rows x 128B)
      __builtin_prefetch(kh + (j0 + 32) * HD + lane * HD, 0, 1);
    v8f L0 = {}, L1 = {};
    const __bf16* kt0 = kh + j0 * HD;
    const __bf16* kt1 = kh + (j0 + 16) * HD;
    L0 = wmma_bf16(qa0, loadB_bf16(kt0, HD, lane), L0);
    L0 = wmma_bf16(qa1, loadB_bf16(kt0 + 32, HD, lane), L0);
    L1 = wmma_bf16(qa0, loadB_bf16(kt1, HD, lane), L1);
    L1 = wmma_bf16(qa1, loadB_bf16(kt1 + 32, HD, lane), L1);
#pragma unroll
    for (int r = 0; r < 8; ++r) {
      const int M = r + 8 * half;
      sL[(M * 32 + nn) * 8 + wid]      = L0[r] * scale;
      sL[(M * 32 + 16 + nn) * 8 + wid] = L1[r] * scale;
    }
    __syncthreads();
#pragma unroll
    for (int t = 0; t < 16; ++t) {                  // row t, column j=lane
      const float* Lp = &sL[(t * 32 + lane) * 8];
      v4f a = *(const v4f*)Lp, b = *(const v4f*)(Lp + 4);
      float lg = wpre_r[0] * a[0] + wpre_r[1] * a[1] + wpre_r[2] * a[2] +
                 wpre_r[3] * a[3] + wpre_r[4] * b[0] + wpre_r[5] * b[1] +
                 wpre_r[6] * b[2] + wpre_r[7] * b[3];
      float nm = fmaxf(rm[t], lg);
      rs[t] = rs[t] * __expf(rm[t] - nm) + __expf(lg - nm);
      rm[t] = nm;
    }
    __syncthreads();
  }
  // merge partial (max,sum) across the 32 lanes (columns j mod 32)
#pragma unroll
  for (int t = 0; t < 16; ++t) {
#pragma unroll
    for (int off = 16; off >= 1; off >>= 1) {
      float om = __shfl_xor(rm[t], off, 32);
      float os = __shfl_xor(rs[t], off, 32);
      float nm = fmaxf(rm[t], om);
      rs[t] = rs[t] * __expf(rm[t] - nm) + os * __expf(om - nm);
      rm[t] = nm;
    }
  }
  if (lane == 0) {
#pragma unroll
    for (int t = 0; t < 16; ++t) {
      sMZ[wid][t][0] = rm[t];
      sMZ[wid][t][1] = 1.0f / rs[t];
    }
  }
  __syncthreads();
  float mM[16], mZ[16];
#pragma unroll
  for (int t = 0; t < 16; ++t) { mM[t] = sMZ[wid][t][0]; mZ[t] = sMZ[wid][t][1]; }

  // ---------------- pass 2: output ----------------
  v8f oacc[4] = {};
  for (int jc = 0; jc < NSEQ / 32; ++jc) {
    const int j0 = jc * 32;
    if (j0 + 32 < NSEQ) {  // prefetch next K chunk + next V chunk
      __builtin_prefetch(kh + (j0 + 32) * HD + lane * HD, 0, 1);
      __builtin_prefetch(vh + (lane * 2) * NSEQ + j0 + 32, 0, 1);
      __builtin_prefetch(vh + (lane * 2 + 1) * NSEQ + j0 + 32, 0, 1);
    }
    v8f L0 = {}, L1 = {};
    const __bf16* kt0 = kh + j0 * HD;
    const __bf16* kt1 = kh + (j0 + 16) * HD;
    L0 = wmma_bf16(qa0, loadB_bf16(kt0, HD, lane), L0);
    L0 = wmma_bf16(qa1, loadB_bf16(kt0 + 32, HD, lane), L0);
    L1 = wmma_bf16(qa0, loadB_bf16(kt1, HD, lane), L1);
    L1 = wmma_bf16(qa1, loadB_bf16(kt1 + 32, HD, lane), L1);
#pragma unroll
    for (int r = 0; r < 8; ++r) {
      const int M = r + 8 * half;
      sL[(M * 32 + nn) * 8 + wid]      = L0[r] * scale;
      sL[(M * 32 + 16 + nn) * 8 + wid] = L1[r] * scale;
    }
    __syncthreads();
    // probabilities of mixed head g=wid
#pragma unroll
    for (int t = 0; t < 16; ++t) {
      const float* Lp = &sL[(t * 32 + lane) * 8];
      v4f a = *(const v4f*)Lp, b = *(const v4f*)(Lp + 4);
      float lg = wpre_r[0] * a[0] + wpre_r[1] * a[1] + wpre_r[2] * a[2] +
                 wpre_r[3] * a[3] + wpre_r[4] * b[0] + wpre_r[5] * b[1] +
                 wpre_r[6] * b[2] + wpre_r[7] * b[3];
      sP[(t * 32 + lane) * 8 + wid] = __expf(lg - mM[t]) * mZ[t];
    }
    __syncthreads();
    // post-mix into WMMA A-operand (bf16), then PV
    v16bf pa;
#pragma unroll
    for (int s = 0; s < 16; ++s) {
      const int j = s + ((s < 8) ? (half ? 8 : 0) : (half ? 16 : 8));
      const float* Pp = &sP[(nn * 32 + j) * 8];     // row m = lane&15
      v4f a = *(const v4f*)Pp, b = *(const v4f*)(Pp + 4);
      float v = wpost_r[0] * a[0] + wpost_r[1] * a[1] + wpost_r[2] * a[2] +
                wpost_r[3] * a[3] + wpost_r[4] * b[0] + wpost_r[5] * b[1] +
                wpost_r[6] * b[2] + wpost_r[7] * b[3];
      pa[s] = (__bf16)v;
    }
#pragma unroll
    for (int t = 0; t < 4; ++t) {
      const __bf16* vb = vh + (t * 16) * NSEQ + j0;  // B col = d, K = j (contig)
      oacc[t] = wmma_bf16(pa, loadB_bf16(vb, NSEQ, lane), oacc[t]);
    }
    // (next iteration's sL write is safe: everyone passed the sync above
    //  before any wave can reach the following sync-protected sP write)
  }

  // ctx[b][n][dim] bf16
#pragma unroll
  for (int t = 0; t < 4; ++t)
#pragma unroll
    for (int r = 0; r < 8; ++r) {
      const int M = r + 8 * half;
      ctx[(bb * NSEQ + i0 + M) * DIM + wid * HD + t * 16 + nn] =
          (__bf16)oacc[t][r];
    }
}

// ===========================================================================
// Kernel 3: out = ctx @ w_out^T + b_out   (f32 output)
// ===========================================================================
__global__ __launch_bounds__(256) void k_outproj(
    const __bf16* __restrict__ ctx, const float* __restrict__ wout,
    const float* __restrict__ bout, float* __restrict__ out) {
  const int lane = threadIdx.x & 31;
  const int w = blockIdx.x * 8 + (threadIdx.x >> 5);  // 2048 waves
  const int rowTile = w >> 3, cgrp = w & 7;
  const __bf16* A0 = ctx + (rowTile * 16) * DIM;
  const float* B0 = wout + (cgrp * 64) * DIM;

  v8f acc[4] = {};
  for (int k = 0; k < DIM; k += 32) {
    if (k + 32 < DIM) {
      __builtin_prefetch(A0 + k + 32 + (lane & 15) * DIM, 0, 1);
      __builtin_prefetch(B0 + k + 32 + lane * DIM, 0, 1);
      __builtin_prefetch(B0 + k + 32 + (lane + 32) * DIM, 0, 1);
    }
    v16bf a = loadA_bf16(A0 + k, DIM, lane);
#pragma unroll
    for (int t = 0; t < 4; ++t)
      acc[t] = wmma_bf16(a, loadB_f32(B0 + (t * 16) * DIM + k, DIM, lane), acc[t]);
  }
  const int half = lane >> 4, nn = lane & 15;
  float bias[4];
#pragma unroll
  for (int t = 0; t < 4; ++t) bias[t] = bout[cgrp * 64 + t * 16 + nn];
#pragma unroll
  for (int t = 0; t < 4; ++t)
#pragma unroll
    for (int r = 0; r < 8; ++r) {
      const int gr = rowTile * 16 + r + 8 * half;
      out[gr * DIM + cgrp * 64 + t * 16 + nn] = acc[t][r] + bias[t];
    }
}

// ===========================================================================
extern "C" void kernel_launch(void* const* d_in, const int* in_sizes, int n_in,
                              void* d_out, int out_size, void* d_ws,
                              size_t ws_size, hipStream_t stream) {
  const float* x    = (const float*)d_in[0];
  const float* wqkv = (const float*)d_in[1];
  const float* bqkv = (const float*)d_in[2];
  const float* wpre = (const float*)d_in[3];
  const float* wpost = (const float*)d_in[4];
  const float* wout = (const float*)d_in[5];
  const float* bout = (const float*)d_in[6];
  float* out = (float*)d_out;

  char* ws = (char*)d_ws;
  __bf16* qb  = (__bf16*)(ws);                       // 4 MB
  __bf16* kb  = (__bf16*)(ws + (4u << 20));          // 4 MB
  __bf16* vT  = (__bf16*)(ws + (8u << 20));          // 4 MB
  __bf16* ctx = (__bf16*)(ws + (12u << 20));         // 4 MB

  // 256 row-tiles x 24 col-groups = 6144 waves / 8 per block
  k_qkv_rope<<<768, 256, 0, stream>>>(x, wqkv, bqkv, qb, kb, vT);
  // 2 batches x 128 query tiles
  k_attn<<<256, 256, 0, stream>>>(qb, kb, vT, wpre, wpost, ctx);
  // 256 row-tiles x 8 col-groups = 2048 waves / 8 per block
  k_outproj<<<256, 256, 0, stream>>>(ctx, wout, bout, out);
}
